// CausalSelfAttention_73229192397504
// MI455X (gfx1250) — matrix operand order repaired and checked by reading, not compiled, not run on hardware
//
#include <hip/hip_runtime.h>

// ---------------------------------------------------------------------------
// CDNA5 (gfx1250, wave32) causal GQA attention block.
// bf16 convert -> WMMA GEMM (q,k,v proj, 64x64 wave tile) -> RoPE+RMSNorm ->
// flash attention (TDM-staged K/V tiles, WMMA QK^T + WMMA PV, online softmax)
// -> WMMA GEMM (Wo proj, fp32 out).
// ---------------------------------------------------------------------------

typedef __bf16 bf16;
typedef __attribute__((ext_vector_type(16))) __bf16 v16bf;
typedef __attribute__((ext_vector_type(8)))  __bf16 v8bf;
typedef __attribute__((ext_vector_type(8)))  float  v8f;

#define B_    2
#define T_    2048
#define C_    1024
#define H_    16
#define KVH_  4
#define D_    64

#if defined(__HIP_DEVICE_COMPILE__) && __has_builtin(__builtin_amdgcn_tensor_load_to_lds) && \
    __has_builtin(__builtin_amdgcn_s_wait_tensorcnt)
#define USE_TDM 1
#else
#define USE_TDM 0
#endif

// ----- WMMA helpers ---------------------------------------------------------

__device__ __forceinline__ v8f wmma_bf16(v16bf a, v16bf b, v8f c) {
    // (neg_a, A, neg_b, B, c_mod, C, reuse_a, reuse_b)
    return __builtin_amdgcn_wmma_f32_16x16x32_bf16(false, a, false, b,
                                                   (short)0, c, false, false);
}

// A fragment: 16x32 bf16 tile, row-major, leading dim ld (elements).
// Lane L: row = L%16; elems 0..7 = K (L/16)*8+{0..7}; elems 8..15 = K 16+(L/16)*8+{0..7}.
__device__ __forceinline__ v16bf load_a(const bf16* base, int ld) {
    const int lane = threadIdx.x & 31;
    const int row  = lane & 15;
    const int half = lane >> 4;
    const bf16* p0 = base + (size_t)row * ld + half * 8;
    v8bf lo = *(const v8bf*)p0;
    v8bf hi = *(const v8bf*)(p0 + 16);
    v16bf r;
#pragma unroll
    for (int i = 0; i < 8; ++i) { r[i] = lo[i]; r[i + 8] = hi[i]; }
    return r;
}

// B fragment for C = A * B^T where B-source is row-major [N, K]:
// 32(K) x 16(N) hardware tile. Lane L: col n = L%16; elems 0..7 = K (L/16)*8+{0..7}; elems 8..15 = K 16+(L/16)*8+{0..7}.
__device__ __forceinline__ v16bf load_b(const bf16* base, int ld) {
    const int lane = threadIdx.x & 31;
    const int row  = lane & 15;
    const int half = lane >> 4;
    const bf16* p = base + (size_t)row * ld + half * 8;
    v8bf lo = *(const v8bf*)p;
    v8bf hi = *(const v8bf*)(p + 16);
    v16bf r;
#pragma unroll
    for (int i = 0; i < 8; ++i) { r[i] = lo[i]; r[i + 8] = hi[i]; }
    return r;
}

// ----- Tensor Data Mover: 2D bf16 tile -> LDS (ISA 08_async_tensor D#) ------

#if USE_TDM
typedef __attribute__((ext_vector_type(4))) unsigned int u32x4;
typedef __attribute__((ext_vector_type(8))) int          i32x8;
typedef __attribute__((ext_vector_type(4))) int          i32x4;

__device__ __forceinline__ void tdm_load_2d_bf16(const void* lds_ptr, const bf16* gptr,
                                                 unsigned int tensor_d0, unsigned int tensor_d1,
                                                 unsigned int tile_d0, unsigned int tile_d1,
                                                 unsigned long long stride0) {
    unsigned long long ga = (unsigned long long)(uintptr_t)gptr;
    unsigned int lds_off = (unsigned int)(uintptr_t)lds_ptr; // low 32b of LDS aperture addr
    u32x4 g0;
    g0[0] = 1u;                                               // count=1, user descriptor
    g0[1] = lds_off;                                          // lds byte address
    g0[2] = (unsigned int)(ga & 0xFFFFFFFFu);                 // global addr [31:0]
    g0[3] = (unsigned int)((ga >> 32) & 0x1FFFFFFu) | (2u << 30); // [56:32] | type=2
    i32x8 g1;
    g1[0] = (int)(1u << 16);                                  // data_size=1 (2 bytes)
    g1[1] = (int)((tensor_d0 & 0xFFFFu) << 16);               // tensor_dim0 lo
    g1[2] = (int)(((tensor_d0 >> 16) & 0xFFFFu) | ((tensor_d1 & 0xFFFFu) << 16));
    g1[3] = (int)(((tensor_d1 >> 16) & 0xFFFFu) | ((tile_d0 & 0xFFFFu) << 16));
    g1[4] = (int)(tile_d1 & 0xFFFFu);                         // tile_dim1 (tile_dim2 = 0)
    g1[5] = (int)(unsigned int)(stride0 & 0xFFFFFFFFull);     // tensor_dim0_stride lo
    g1[6] = (int)(unsigned int)((stride0 >> 32) & 0xFFFFull); // stride hi (dim1_stride=0)
    g1[7] = 0;
    i32x4 z4 = {0, 0, 0, 0};
    i32x8 z8 = {0, 0, 0, 0, 0, 0, 0, 0};
    // 6-arg toolchain variant: (g0, g1, g2, g3, g4, cpol)
    __builtin_amdgcn_tensor_load_to_lds(g0, g1, z4, z4, z8, 0);
}
#endif

// ----- elementwise convert --------------------------------------------------

__global__ void cvt_f32_bf16(const float* __restrict__ in, bf16* __restrict__ out, int n) {
    int i = blockIdx.x * blockDim.x + threadIdx.x;
    if (i < n) out[i] = (bf16)in[i];
}

// ----- generic WMMA GEMM: C[M,N] = A[M,K] * W[N,K]^T ------------------------
// block = 128 threads (4 waves, 2x2), block tile 128x128, wave tile 64x64
// (4x4 fragments -> 16 WMMA per 16 b128 loads per K-step).

__global__ void __launch_bounds__(128, 1)
gemm_bf16_wmma(const bf16* __restrict__ A, const bf16* __restrict__ W,
               float* __restrict__ C, int M, int N, int K) {
    const int lane = threadIdx.x & 31;
    const int wave = threadIdx.x >> 5;
    const int m0 = blockIdx.y * 128 + (wave >> 1) * 64;
    const int n0 = blockIdx.x * 128 + (wave & 1) * 64;
    const int col  = lane & 15;
    const int half = lane >> 4;

    v8f acc[4][4] = {};
    for (int k0 = 0; k0 < K; k0 += 32) {
        v16bf b[4];
#pragma unroll
        for (int nf = 0; nf < 4; ++nf)
            b[nf] = load_b(W + (size_t)(n0 + nf * 16) * K + k0, K);
#pragma unroll
        for (int mf = 0; mf < 4; ++mf) {
            v16bf a = load_a(A + (size_t)(m0 + mf * 16) * K + k0, K);
#pragma unroll
            for (int nf = 0; nf < 4; ++nf)
                acc[mf][nf] = wmma_bf16(a, b[nf], acc[mf][nf]);
        }
    }
#pragma unroll
    for (int mf = 0; mf < 4; ++mf)
#pragma unroll
        for (int nf = 0; nf < 4; ++nf)
#pragma unroll
            for (int r = 0; r < 8; ++r) {
                int mm = m0 + mf * 16 + half * 8 + r;
                int nn = n0 + nf * 16 + col;
                C[(size_t)mm * N + nn] = acc[mf][nf][r];
            }
}

// ----- RoPE + RMSNorm + layout shuffle --------------------------------------
// grid.x = B*T ; grid.y = 24 (16 q heads, 4 k heads, 4 v heads) ; block = 32.
// Outputs: Qn [B,H,T,64] bf16 ; Kn [B,KVH,T,64] bf16 ; Vt [B,KVH,64,T] bf16.

__global__ void rope_rms(const float* __restrict__ qf, const float* __restrict__ kf,
                         const float* __restrict__ vf, const float* __restrict__ cosp,
                         const float* __restrict__ sinp,
                         bf16* __restrict__ Qn, bf16* __restrict__ Kn, bf16* __restrict__ Vt) {
    const int bt = blockIdx.x;
    const int b  = bt / T_;
    const int t  = bt % T_;
    const int y  = blockIdx.y;
    const int j  = threadIdx.x; // 0..31

    if (y < 20) {
        const bool isq = (y < 16);
        const int h = isq ? y : (y - 16);
        const float* src = isq ? (qf + (size_t)bt * C_ + h * D_)
                               : (kf + (size_t)bt * (KVH_ * D_) + h * D_);
        float x1 = src[j], x2 = src[j + 32];
        float c = cosp[t * 32 + j], s = sinp[t * 32 + j];
        float r1 =  x1 * c + x2 * s;
        float r2 = -x1 * s + x2 * c;
        float ss = r1 * r1 + r2 * r2;
#pragma unroll
        for (int off = 16; off >= 1; off >>= 1) ss += __shfl_xor(ss, off, 32);
        float inv = rsqrtf(ss * (1.0f / 64.0f) + 1.1920929e-07f);
        bf16* dst = isq ? (Qn + (((size_t)b * H_   + h) * T_ + t) * D_)
                        : (Kn + (((size_t)b * KVH_ + h) * T_ + t) * D_);
        dst[j]      = (bf16)(r1 * inv);
        dst[j + 32] = (bf16)(r2 * inv);
    } else {
        const int h = y - 20;
        const float* src = vf + (size_t)bt * (KVH_ * D_) + h * D_;
        bf16* dst = Vt + ((size_t)b * KVH_ + h) * D_ * (size_t)T_;
        dst[(size_t)j        * T_ + t] = (bf16)src[j];
        dst[(size_t)(j + 32) * T_ + t] = (bf16)src[j + 32];
    }
}

// ----- flash attention (one wave = 32 q rows) -------------------------------
// grid = (T/128, H, B), block = 128 (4 waves). att out: [B,T,H*D] bf16.
// Per-wave LDS: K tile 32x64 (4KB) + V tile 64x32 (4KB) + P tile 32x32 (2KB).

#define SMEM_PER_WAVE (4096 + 4096 + 2048)

__global__ void attn_wmma(const bf16* __restrict__ Qn, const bf16* __restrict__ Kn,
                          const bf16* __restrict__ Vt, bf16* __restrict__ att) {
    __shared__ __align__(16) char smem[4 * SMEM_PER_WAVE]; // 40 KB of 320 KB

    const int lane = threadIdx.x & 31;
    const int wave = threadIdx.x >> 5;
    const int col  = lane & 15;
    const int half = lane >> 4;

    const int b = blockIdx.z;
    const int h = blockIdx.y;
    const int q0 = blockIdx.x * 128 + wave * 32;
    const int kvh = h >> 2; // N_HEAD / N_KV_HEAD = 4

    const bf16* Qh = Qn + (((size_t)b * H_   + h)   * T_) * D_;
    const bf16* Kh = Kn + (((size_t)b * KVH_ + kvh) * T_) * D_;
    const bf16* Vh = Vt + (((size_t)b * KVH_ + kvh) * D_) * (size_t)T_;

    char* mine = smem + wave * SMEM_PER_WAVE;
    bf16* ldsK = (bf16*)mine;            // [32][64]
    bf16* ldsV = (bf16*)(mine + 4096);   // [64][32]
    bf16* ldsP = (bf16*)(mine + 8192);   // [32][32]

    // Q fragments are k-loop invariant: hoist. aq[qf][d0/32]
    v16bf aq[2][2];
#pragma unroll
    for (int qf = 0; qf < 2; ++qf)
#pragma unroll
        for (int dd = 0; dd < 2; ++dd)
            aq[qf][dd] = load_a(Qh + (size_t)(q0 + qf * 16) * D_ + dd * 32, D_);

    float m_i[2][8], l_i[2][8];
    v8f acc[2][4] = {};
#pragma unroll
    for (int qf = 0; qf < 2; ++qf)
#pragma unroll
        for (int r = 0; r < 8; ++r) { m_i[qf][r] = -3.0e38f; l_i[qf][r] = 0.0f; }

    for (int kb = 0; kb <= q0; kb += 32) {
#if USE_TDM
        // Async TDM: K tile first, V tile second (in-order per wave).
        tdm_load_2d_bf16(ldsK, Kh + (size_t)kb * D_, D_, T_, D_, 32, D_);
        tdm_load_2d_bf16(ldsV, Vh + kb, T_, D_, 32, D_, T_);
#else
        if (kb + 32 <= q0)
            __builtin_prefetch(Kh + (size_t)(kb + 32) * D_ + lane * 4, 0, 0);
#endif
        // --- S = Q K^T over head_dim=64 (two K-steps of 32) ---
        v8f s[2][2] = {};
#if USE_TDM
        __builtin_amdgcn_s_wait_tensorcnt((short)1); // K tile ready; V still in flight
#endif
#pragma unroll
        for (int dd = 0; dd < 2; ++dd) {
            const int d0 = dd * 32;
#if USE_TDM
            v16bf bk0 = load_b(ldsK + d0, D_);
            v16bf bk1 = load_b(ldsK + 16 * D_ + d0, D_);
#else
            v16bf bk0 = load_b(Kh + (size_t)kb * D_ + d0, D_);
            v16bf bk1 = load_b(Kh + (size_t)(kb + 16) * D_ + d0, D_);
#endif
            s[0][0] = wmma_bf16(aq[0][dd], bk0, s[0][0]);
            s[0][1] = wmma_bf16(aq[0][dd], bk1, s[0][1]);
            s[1][0] = wmma_bf16(aq[1][dd], bk0, s[1][0]);
            s[1][1] = wmma_bf16(aq[1][dd], bk1, s[1][1]);
        }
        // --- scale + causal mask (diagonal block only) ---
        const bool diag = (kb == q0);
#pragma unroll
        for (int qf = 0; qf < 2; ++qf)
#pragma unroll
            for (int kf = 0; kf < 2; ++kf)
#pragma unroll
                for (int r = 0; r < 8; ++r) {
                    float v = s[qf][kf][r] * 0.125f; // 1/sqrt(64)
                    if (diag) {
                        int qr = q0 + qf * 16 + half * 8 + r;
                        int kc = kb + kf * 16 + col;
                        if (kc > qr) v = -3.0e38f;
                    }
                    s[qf][kf][r] = v;
                }
        // --- online softmax; P -> LDS (bf16) ---
#pragma unroll
        for (int qf = 0; qf < 2; ++qf) {
            float mnew[8], sc[8];
#pragma unroll
            for (int r = 0; r < 8; ++r) {
                float v = fmaxf(s[qf][0][r], s[qf][1][r]);
#pragma unroll
                for (int off = 8; off >= 1; off >>= 1) v = fmaxf(v, __shfl_xor(v, off, 32));
                mnew[r] = fmaxf(m_i[qf][r], v);
                sc[r]   = __expf(m_i[qf][r] - mnew[r]);
                m_i[qf][r] = mnew[r];
            }
#pragma unroll
            for (int kf = 0; kf < 2; ++kf)
#pragma unroll
                for (int r = 0; r < 8; ++r)
                    s[qf][kf][r] = __expf(s[qf][kf][r] - mnew[r]);
#pragma unroll
            for (int r = 0; r < 8; ++r) {
                float v = s[qf][0][r] + s[qf][1][r];
#pragma unroll
                for (int off = 8; off >= 1; off >>= 1) v += __shfl_xor(v, off, 32);
                l_i[qf][r] = l_i[qf][r] * sc[r] + v;
            }
#pragma unroll
            for (int df = 0; df < 4; ++df)
#pragma unroll
                for (int r = 0; r < 8; ++r) acc[qf][df][r] *= sc[r];
#pragma unroll
            for (int kf = 0; kf < 2; ++kf)
#pragma unroll
                for (int r = 0; r < 8; ++r)
                    ldsP[(qf * 16 + half * 8 + r) * 32 + kf * 16 + col] = (bf16)s[qf][kf][r];
        }
        // --- O += P V (contraction over 32 keys) ---
        v16bf pa0 = load_a(ldsP, 32);
        v16bf pa1 = load_a(ldsP + 16 * 32, 32);
#if USE_TDM
        __builtin_amdgcn_s_wait_tensorcnt((short)0); // V tile ready
#endif
#pragma unroll
        for (int df = 0; df < 4; ++df) {
#if USE_TDM
            v16bf bv = load_b(ldsV + (size_t)(df * 16) * 32, 32);
#else
            v16bf bv = load_b(Vh + (size_t)(df * 16) * T_ + kb, T_);
#endif
            acc[0][df] = wmma_bf16(pa0, bv, acc[0][df]);
            acc[1][df] = wmma_bf16(pa1, bv, acc[1][df]);
        }
    }
    // --- epilogue: normalize, write bf16 att[B,T,H*D] ---
#pragma unroll
    for (int qf = 0; qf < 2; ++qf)
#pragma unroll
        for (int df = 0; df < 4; ++df)
#pragma unroll
            for (int r = 0; r < 8; ++r) {
                int tq = q0 + qf * 16 + half * 8 + r;
                float o = acc[qf][df][r] / l_i[qf][r];
                att[((size_t)b * T_ + tq) * C_ + h * D_ + df * 16 + col] = (bf16)o;
            }
}

// ---------------------------------------------------------------------------

extern "C" void kernel_launch(void* const* d_in, const int* in_sizes, int n_in,
                              void* d_out, int out_size, void* d_ws, size_t ws_size,
                              hipStream_t stream) {
    const float* x    = (const float*)d_in[0];
    const float* cosp = (const float*)d_in[1];
    const float* sinp = (const float*)d_in[2];
    const float* Wq   = (const float*)d_in[3];
    const float* Wk   = (const float*)d_in[4];
    const float* Wv   = (const float*)d_in[5];
    const float* Wo   = (const float*)d_in[6];
    float* out = (float*)d_out;

    const size_t BT = (size_t)B_ * T_;           // 4096
    const size_t nX  = BT * C_;                  // 4,194,304
    const size_t nWq = (size_t)C_ * C_;          // 1,048,576
    const size_t nWk = (size_t)(KVH_ * D_) * C_; // 262,144
    const size_t nQ  = BT * C_;
    const size_t nK  = BT * (KVH_ * D_);

    char* ws = (char*)d_ws;
    size_t off = 0;
    auto carve = [&](size_t bytes) { char* p = ws + off; off += (bytes + 255) & ~(size_t)255; return p; };

    bf16*  xb   = (bf16*)carve(nX  * 2);
    bf16*  wqb  = (bf16*)carve(nWq * 2);
    bf16*  wkb  = (bf16*)carve(nWk * 2);
    bf16*  wvb  = (bf16*)carve(nWk * 2);
    bf16*  wob  = (bf16*)carve(nWq * 2);
    float* qf32 = (float*)carve(nQ * 4);
    float* kf32 = (float*)carve(nK * 4);
    float* vf32 = (float*)carve(nK * 4);
    bf16*  Qn   = (bf16*)carve(nQ * 2);
    bf16*  Kn   = (bf16*)carve(nK * 2);
    bf16*  Vt   = (bf16*)carve(nK * 2);
    bf16*  attb = (bf16*)carve(nQ * 2);
    (void)ws_size; (void)in_sizes; (void)n_in; (void)out_size;

    // 1) fp32 -> bf16 conversions
    auto cvt = [&](const float* src, bf16* dst, size_t n) {
        cvt_f32_bf16<<<(unsigned)((n + 255) / 256), 256, 0, stream>>>(src, dst, (int)n);
    };
    cvt(x, xb, nX);
    cvt(Wq, wqb, nWq);
    cvt(Wk, wkb, nWk);
    cvt(Wv, wvb, nWk);
    cvt(Wo, wob, nWq);

    // 2) QKV projections: [4096, K=1024] x [N, 1024]^T ; block tile 128x128
    gemm_bf16_wmma<<<dim3(C_ / 128, (unsigned)(BT / 128)), 128, 0, stream>>>(
        xb, wqb, qf32, (int)BT, C_, C_);
    gemm_bf16_wmma<<<dim3((KVH_ * D_) / 128, (unsigned)(BT / 128)), 128, 0, stream>>>(
        xb, wkb, kf32, (int)BT, KVH_ * D_, C_);
    gemm_bf16_wmma<<<dim3((KVH_ * D_) / 128, (unsigned)(BT / 128)), 128, 0, stream>>>(
        xb, wvb, vf32, (int)BT, KVH_ * D_, C_);

    // 3) RoPE + RMSNorm + relayout (+ V transpose)
    rope_rms<<<dim3((unsigned)BT, H_ + KVH_ + KVH_), 32, 0, stream>>>(
        qf32, kf32, vf32, cosp, sinp, Qn, Kn, Vt);

    // 4) flash attention (TDM-staged K/V)
    attn_wmma<<<dim3(T_ / 128, H_, B_), 128, 0, stream>>>(Qn, Kn, Vt, attb);

    // 5) output projection -> fp32 d_out
    gemm_bf16_wmma<<<dim3(C_ / 128, (unsigned)(BT / 128)), 128, 0, stream>>>(
        attb, wob, out, (int)BT, C_, C_);
}
